// SpatialAttentionLayer_28595892257664
// MI455X (gfx1250) — compile-verified, matrix operand order checked by reading, not run on hardware
//
#include <hip/hip_runtime.h>
#include <hip/hip_bf16.h>

typedef __attribute__((ext_vector_type(16))) __bf16          v16bf;
typedef __attribute__((ext_vector_type(8)))  float           v8f;
typedef __attribute__((ext_vector_type(16))) unsigned short  v16us;
typedef __attribute__((ext_vector_type(8)))  unsigned short  v8us;

#define NN 8192
#define DD 256

// f32 -> bf16 round-to-nearest-even
__device__ __forceinline__ unsigned short f2bf(float f) {
  unsigned int u = __builtin_bit_cast(unsigned int, f);
  u += 0x7FFFu + ((u >> 16) & 1u);
  return (unsigned short)(u >> 16);
}

// Load one 16x32 bf16 WMMA operand fragment (A-row or B-column form).
// p points at the 32-element K segment for this lane's row/col; khi = lane>>4.
// Per ISA: elems 0..7 = K[khi*8 .. +7], elems 8..15 = K[16+khi*8 .. +7].
__device__ __forceinline__ v16bf frag_ld(const unsigned short* p, int khi) {
  v8us x0 = *(const v8us*)(p + khi * 8);
  v8us x1 = *(const v8us*)(p + 16 + khi * 8);
  v16us u;
#pragma unroll
  for (int i = 0; i < 8; ++i) { u[i] = x0[i]; u[i + 8] = x1[i]; }
  return __builtin_bit_cast(v16bf, u);
}

// ---------------------------------------------------------------------------
// Kernel 1: one projection (templated, branch-free stores). grid = N/128.
//   WHICH==0/1: row-major bf16 out [N][256]   WHICH==2: transposed out [256][N]
// ---------------------------------------------------------------------------
template <int WHICH>
__global__ __launch_bounds__(256) void proj_kernel(
    const float* __restrict__ X, const float* __restrict__ W,
    const float* __restrict__ bias, unsigned short* __restrict__ dst) {
  const int tid  = threadIdx.x;
  const int wave = tid >> 5, lane = tid & 31;
  const int hi = lane >> 4, nlo = lane & 15;
  const int rowBase = blockIdx.x * 128 + wave * 16;

  __shared__ __align__(16) unsigned short Wt[64][264];  // W^T panel, bf16
  __shared__ float bsh[64];

  for (int panel = 0; panel < 4; ++panel) {
    const int colBase = panel * 64;
    __syncthreads();
    for (int idx = tid; idx < 64 * 256; idx += 256) {
      int kk = idx >> 6, nn = idx & 63;
      Wt[nn][kk] = f2bf(W[kk * DD + colBase + nn]);
    }
    if (tid < 64) bsh[tid] = bias[colBase + tid];
    __syncthreads();

    v8f acc[4];
#pragma unroll
    for (int cb = 0; cb < 4; ++cb)
#pragma unroll
      for (int i = 0; i < 8; ++i) acc[cb][i] = 0.0f;

    for (int kb = 0; kb < 8; ++kb) {
      const int k = kb * 32;
      const float* xp = X + (size_t)(rowBase + nlo) * DD + k;
      v16us ua;
#pragma unroll
      for (int i = 0; i < 8; ++i) {
        ua[i]     = f2bf(xp[hi * 8 + i]);
        ua[i + 8] = f2bf(xp[16 + hi * 8 + i]);
      }
      v16bf a = __builtin_bit_cast(v16bf, ua);
#pragma unroll
      for (int cb = 0; cb < 4; ++cb) {
        v16bf b = frag_ld(&Wt[cb * 16 + nlo][0] + k, hi);
        acc[cb] = __builtin_amdgcn_wmma_f32_16x16x32_bf16(
            false, a, false, b, (short)0, acc[cb], false, false);
      }
    }
#pragma unroll
    for (int cb = 0; cb < 4; ++cb) {
      const int col = colBase + cb * 16 + nlo;
      const float bb = bsh[cb * 16 + nlo];
#pragma unroll
      for (int g = 0; g < 8; ++g) {
        const int row = rowBase + g + 8 * hi;
        unsigned short h = f2bf(acc[cb][g] + bb);
        if (WHICH == 2) dst[(size_t)col * NN + row] = h;   // compile-time branch
        else            dst[(size_t)row * DD + col] = h;
      }
    }
  }
}

// ---------------------------------------------------------------------------
// Kernel 2: fused spatial attention, flash style. grid = N/128, 256 threads.
// Each wave owns 16 query rows; KV tiles of 64. Online softmax-one.
// ---------------------------------------------------------------------------
__global__ __launch_bounds__(256) void spatial_attn_kernel(
    const unsigned short* __restrict__ Qg, const unsigned short* __restrict__ Kg,
    const unsigned short* __restrict__ Vtg,
    const float* __restrict__ P, const float* __restrict__ X,
    float* __restrict__ Out) {
  const int tid  = threadIdx.x;
  const int wave = tid >> 5, lane = tid & 31;
  const int hi = lane >> 4, nlo = lane & 15;
  const int rowBase = blockIdx.x * 128 + wave * 16;

  // per-wave probability slab: C-layout -> A-layout transpose via LDS
  __shared__ __align__(16) unsigned short Plds[8][16][72];

  // positions of this lane's 8 C-layout rows
  float Pi[8][3];
#pragma unroll
  for (int g = 0; g < 8; ++g) {
    const int r = rowBase + g + 8 * hi;
#pragma unroll
    for (int d = 0; d < 3; ++d) Pi[g][d] = P[r * 3 + d];
  }

  v8f accO[16];
#pragma unroll
  for (int cb = 0; cb < 16; ++cb)
#pragma unroll
    for (int i = 0; i < 8; ++i) accO[cb][i] = 0.0f;
  float rowM[8], rowL[8];
#pragma unroll
  for (int g = 0; g < 8; ++g) { rowM[g] = -3.0e38f; rowL[g] = 0.0f; }

  for (int t = 0; t < NN / 64; ++t) {
    const int kvBase = t * 64;
    if (t + 1 < NN / 64)
      __builtin_prefetch(Kg + (size_t)(kvBase + 64 + nlo) * DD, 0, 1);

    // ---- S = (Q K^T)/16 : wave computes 16x64
    v8f s[4];
#pragma unroll
    for (int nb = 0; nb < 4; ++nb)
#pragma unroll
      for (int i = 0; i < 8; ++i) s[nb][i] = 0.0f;

    for (int kb = 0; kb < 8; ++kb) {
      const int k = kb * 32;
      v16bf a = frag_ld(Qg + (size_t)(rowBase + nlo) * DD + k, hi);
#pragma unroll
      for (int nb = 0; nb < 4; ++nb) {
        v16bf b = frag_ld(Kg + (size_t)(kvBase + nb * 16 + nlo) * DD + k, hi);
        s[nb] = __builtin_amdgcn_wmma_f32_16x16x32_bf16(
            false, a, false, b, (short)0, s[nb], false, false);
      }
    }

    // ---- spatial decay + online softmax-one statistics
    float Pj[4][3];
#pragma unroll
    for (int nb = 0; nb < 4; ++nb) {
      const int j = kvBase + nb * 16 + nlo;
#pragma unroll
      for (int d = 0; d < 3; ++d) Pj[nb][d] = P[j * 3 + d];
    }

    float pv[4][8];
    float corr[8];
#pragma unroll
    for (int g = 0; g < 8; ++g) {
      float mx = -3.0e38f;
#pragma unroll
      for (int nb = 0; nb < 4; ++nb) {
        float dx = Pi[g][0] - Pj[nb][0];
        float dy = Pi[g][1] - Pj[nb][1];
        float dz = Pi[g][2] - Pj[nb][2];
        float dist2 = dx * dx + dy * dy + dz * dz;
        float sv = s[nb][g] * 0.0625f * __expf(-0.5f * dist2);
        pv[nb][g] = sv;
        mx = fmaxf(mx, sv);
      }
#pragma unroll
      for (int m = 1; m < 16; m <<= 1) mx = fmaxf(mx, __shfl_xor(mx, m, 32));
      float newm = fmaxf(rowM[g], mx);
      corr[g] = __expf(rowM[g] - newm);
      rowM[g] = newm;
      float ls = 0.0f;
#pragma unroll
      for (int nb = 0; nb < 4; ++nb) {
        float e = __expf(pv[nb][g] - newm);
        pv[nb][g] = e;
        ls += e;
      }
#pragma unroll
      for (int m = 1; m < 16; m <<= 1) ls += __shfl_xor(ls, m, 32);
      rowL[g] = rowL[g] * corr[g] + ls;
    }
#pragma unroll
    for (int cb = 0; cb < 16; ++cb)
#pragma unroll
      for (int g = 0; g < 8; ++g) accO[cb][g] *= corr[g];

    // stash probabilities (bf16) — wave-private slab, no barrier needed
#pragma unroll
    for (int g = 0; g < 8; ++g)
#pragma unroll
      for (int nb = 0; nb < 4; ++nb)
        Plds[wave][g + 8 * hi][nb * 16 + nlo] = f2bf(pv[nb][g]);

    // ---- O += A @ V  (B columns come contiguous from V^T)
#pragma unroll
    for (int ks = 0; ks < 2; ++ks) {
      const int k = ks * 32;
      v16bf a = frag_ld(&Plds[wave][nlo][0] + k, hi);
#pragma unroll
      for (int cb = 0; cb < 16; ++cb) {
        v16bf b = frag_ld(Vtg + (size_t)(cb * 16 + nlo) * NN + kvBase + k, hi);
        accO[cb] = __builtin_amdgcn_wmma_f32_16x16x32_bf16(
            false, a, false, b, (short)0, accO[cb], false, false);
      }
    }
  }

  // epilogue: softmax-one denominator (1 + l), residual, f32 store
#pragma unroll
  for (int g = 0; g < 8; ++g) rowL[g] = 1.0f / (1.0f + rowL[g]);
#pragma unroll
  for (int cb = 0; cb < 16; ++cb) {
    const int col = cb * 16 + nlo;
#pragma unroll
    for (int g = 0; g < 8; ++g) {
      const int row = rowBase + g + 8 * hi;
      Out[(size_t)row * DD + col] =
          accO[cb][g] * rowL[g] + X[(size_t)row * DD + col];
    }
  }
}

extern "C" void kernel_launch(void* const* d_in, const int* in_sizes, int n_in,
                              void* d_out, int out_size, void* d_ws, size_t ws_size,
                              hipStream_t stream) {
  const float* X  = (const float*)d_in[0];
  const float* P  = (const float*)d_in[1];
  const float* Wq = (const float*)d_in[2];
  const float* bq = (const float*)d_in[3];
  const float* Wk = (const float*)d_in[4];
  const float* bk = (const float*)d_in[5];
  const float* Wv = (const float*)d_in[6];
  const float* bv = (const float*)d_in[7];
  float* Out = (float*)d_out;

  unsigned short* Qg  = (unsigned short*)d_ws;           // N*D bf16 = 4 MB
  unsigned short* Kg  = Qg + (size_t)NN * DD;            // 4 MB
  unsigned short* Vtg = Kg + (size_t)NN * DD;            // V^T, 4 MB

  dim3 blk(256), grd(NN / 128);
  proj_kernel<0><<<grd, blk, 0, stream>>>(X, Wq, bq, Qg);
  proj_kernel<1><<<grd, blk, 0, stream>>>(X, Wk, bk, Kg);
  proj_kernel<2><<<grd, blk, 0, stream>>>(X, Wv, bv, Vtg);
  spatial_attn_kernel<<<grd, blk, 0, stream>>>(Qg, Kg, Vtg, P, X, Out);
}
